// GINet_74706661146998
// MI455X (gfx1250) — matrix operand order repaired and checked by reading, not compiled, not run on hardware
//
#include <hip/hip_runtime.h>
#include <hip/hip_bf16.h>
#include <math.h>

typedef __bf16 bf16_t;
typedef __attribute__((ext_vector_type(16))) __bf16 v16bf;
typedef __attribute__((ext_vector_type(8)))  __bf16 bf16x8;
typedef __attribute__((ext_vector_type(8)))  float   v8f;
typedef __attribute__((ext_vector_type(4)))  int     v4i;

#ifndef USE_ASYNC_LDS
#if defined(__has_builtin) && __has_builtin(__builtin_amdgcn_global_load_async_to_lds_b128)
#define USE_ASYNC_LDS 1
#else
#define USE_ASYNC_LDS 0
#endif
#endif

#if USE_ASYNC_LDS
typedef __attribute__((address_space(1))) v4i glob_v4i;  // global (AS1)
typedef __attribute__((address_space(3))) v4i lds_v4i;   // LDS (AS3)
#endif

#define NUM_LAYER 5
#define EMB     300
#define EMBP    320     // 10 * 32
#define HID     600
#define HIDP    640     // 20 * 32
#define FEAT    512
#define N_NODES 100000
#define MPNODES 100096  // 782 * 128
#define N_EDGES 200000
#define N_GRAPHS 5000
#define MPG     5120    // 40 * 128
#define N_MOTIF (N_GRAPHS * 8)
#define EPSV    1e-5f
#define LDSPAD  8       // bf16 elements of padding per LDS B row (16B, keeps b128 align)

// ---------------------------------------------------------------- elementwise

__global__ void k_zero_f32(float* p, size_t n) {
  size_t i = (size_t)blockIdx.x * blockDim.x + threadIdx.x;
  if (i < n) p[i] = 0.f;
}
__global__ void k_zero_i32(int* p, size_t n) {
  size_t i = (size_t)blockIdx.x * blockDim.x + threadIdx.x;
  if (i < n) p[i] = 0;
}

__global__ void k_embed(const int* __restrict__ nx, const float* __restrict__ e1,
                        const float* __restrict__ e2, float* __restrict__ H) {
  size_t idx = (size_t)blockIdx.x * blockDim.x + threadIdx.x;
  if (idx >= (size_t)N_NODES * EMBP) return;
  int i = (int)(idx / EMBP), j = (int)(idx % EMBP);
  float v = 0.f;
  if (j < EMB) v = e1[nx[2 * i] * EMB + j] + e2[nx[2 * i + 1] * EMB + j];
  H[idx] = v;
}

// agg = h + (ee1[4] + ee2[0])  (self-loop constant); pad columns stay zero
__global__ void k_init_agg(const float* __restrict__ H, const float* __restrict__ e1,
                           const float* __restrict__ e2, float* __restrict__ AGG) {
  size_t idx = (size_t)blockIdx.x * blockDim.x + threadIdx.x;
  if (idx >= (size_t)N_NODES * EMBP) return;
  int j = (int)(idx % EMBP);
  float v = H[idx];
  if (j < EMB) v += e1[4 * EMB + j] + e2[j];
  else v = 0.f;
  AGG[idx] = v;
}

// scatter-add over edges: agg[dst] += h[src] + ee1[a0] + ee2[a1]
__global__ void k_scatter(const int* __restrict__ ei, const int* __restrict__ ea,
                          const float* __restrict__ e1, const float* __restrict__ e2,
                          const float* __restrict__ H, float* __restrict__ AGG) {
  int e = blockIdx.x;
  int j = blockIdx.y * blockDim.x + threadIdx.x;
  if (j >= EMB) return;
  int s = ei[e], d = ei[N_EDGES + e];
  int a0 = ea[2 * e], a1 = ea[2 * e + 1];
  float v = H[(size_t)s * EMBP + j] + e1[a0 * EMB + j] + e2[a1 * EMB + j];
  atomicAdd(&AGG[(size_t)d * EMBP + j], v);
}

// convert f32 (rows x cols, ld=lds) -> zero-padded bf16 (rowsPad x ldd)
__global__ void k_to_bf16(bf16_t* __restrict__ dst, const float* __restrict__ src,
                          int rows, int cols, int lds, int ldd, int rowsPad) {
  size_t idx = (size_t)blockIdx.x * blockDim.x + threadIdx.x;
  size_t total = (size_t)rowsPad * ldd;
  if (idx >= total) return;
  int r = (int)(idx / ldd), c = (int)(idx % ldd);
  float v = (r < rows && c < cols) ? src[(size_t)r * lds + c] : 0.f;
  dst[idx] = (bf16_t)v;
}

// transpose weight W[K x N] (row-major) into bf16 WT[Np x Kp] (column n contiguous in K)
__global__ void k_wprep(bf16_t* __restrict__ dst, const float* __restrict__ W,
                        int K, int N, int Kp, int Np) {
  size_t idx = (size_t)blockIdx.x * blockDim.x + threadIdx.x;
  size_t total = (size_t)Np * Kp;
  if (idx >= total) return;
  int n = (int)(idx / Kp), k = (int)(idx % Kp);
  float v = (n < N && k < K) ? W[(size_t)k * N + n] : 0.f;
  dst[idx] = (bf16_t)v;
}

// ---------------------------------------------------------------- WMMA GEMM
// C[M x N] = A[M x Kp](bf16, padded rows) * W[Kp x N] + epilogue
// grid = (ceil(M/128), ceil(N/64)), block = 256 (8 waves). Wave -> 16x64 tile.
// The block's 64-column B panel is staged once into LDS (async load-to-LDS on
// CDNA5) and shared by all 8 waves; A streams from global.
enum { EPI_NONE = 0, EPI_RELU = 1, EPI_SOFTPLUS = 2 };

template <int EPI>
__global__ __launch_bounds__(256)
void k_gemm_wmma(const bf16_t* __restrict__ A, int lda,
                 const bf16_t* __restrict__ WT, int Kp,
                 float* __restrict__ C, int ldc,
                 const float* __restrict__ bias, float biasScale,
                 const float* __restrict__ extra, int M, int N) {
  extern __shared__ __align__(16) bf16_t smem[];  // 64 x (Kp + LDSPAD) bf16
  const int KpPad = Kp + LDSPAD;

  // ---- stage B panel (rows n0..n0+63 of WT) into LDS, 16B chunks ----
  const int n0 = blockIdx.y * 64;
  {
    const int nb = Kp >> 3;      // 16B chunks per row
    const int total = nb << 6;   // 64 rows
    for (int c = threadIdx.x; c < total; c += 256) {
      int n = c / nb;
      int k8 = (c - n * nb) << 3;
      const bf16_t* gp = WT + (size_t)(n0 + n) * Kp + k8;
      bf16_t* lp = smem + (size_t)n * KpPad + k8;
#if USE_ASYNC_LDS
      __builtin_amdgcn_global_load_async_to_lds_b128((glob_v4i*)gp, (lds_v4i*)lp, 0, 0);
#else
      *(bf16x8*)lp = *(const bf16x8*)gp;
#endif
    }
#if USE_ASYNC_LDS
#if defined(__has_builtin) && __has_builtin(__builtin_amdgcn_s_wait_asynccnt)
    __builtin_amdgcn_s_wait_asynccnt(0);
#else
    asm volatile("s_wait_asynccnt 0" ::: "memory");
#endif
#endif
    __syncthreads();
  }

  const int lane = threadIdx.x & 31;
  const int wave = threadIdx.x >> 5;
  const int m0 = blockIdx.x * 128 + wave * 16;
  const int mA = m0 + (lane & 15);
  const int kA = (lane >> 4) * 16;   // A frag: K chunk [kA..kA+15]
  const int kB = (lane >> 4) * 16;   // B frag: K chunk [kB..kB+15] of column n
  const int nc = lane & 15;

  v8f acc[4] = {{}, {}, {}, {}};
  const bf16_t* aRow = A + (size_t)mA * lda + kA;
  const bf16_t* bCol = smem + (size_t)nc * KpPad + kB;

  for (int kk = 0; kk < Kp; kk += 32) {
    v16bf a;
    ((bf16x8*)&a)[0] = *(const bf16x8*)(aRow + kk);
    ((bf16x8*)&a)[1] = *(const bf16x8*)(aRow + kk + 8);
    v16bf b0 = *(const v16bf*)(bCol + kk);
    v16bf b1 = *(const v16bf*)(bCol + (size_t)16 * KpPad + kk);
    v16bf b2 = *(const v16bf*)(bCol + (size_t)32 * KpPad + kk);
    v16bf b3 = *(const v16bf*)(bCol + (size_t)48 * KpPad + kk);
    acc[0] = __builtin_amdgcn_wmma_f32_16x16x32_bf16(false, a, false, b0, (short)0, acc[0], false, false);
    acc[1] = __builtin_amdgcn_wmma_f32_16x16x32_bf16(false, a, false, b1, (short)0, acc[1], false, false);
    acc[2] = __builtin_amdgcn_wmma_f32_16x16x32_bf16(false, a, false, b2, (short)0, acc[2], false, false);
    acc[3] = __builtin_amdgcn_wmma_f32_16x16x32_bf16(false, a, false, b3, (short)0, acc[3], false, false);
  }

  const int mr = m0 + ((lane >> 4) << 3);  // C layout: VGPR r -> row mr + r
#pragma unroll
  for (int t = 0; t < 4; ++t) {
    int n = n0 + t * 16 + nc;
    if (n >= N) continue;
    float bn = bias ? bias[n] * biasScale : 0.f;
    float ex = extra ? extra[n] : 0.f;
#pragma unroll
    for (int r = 0; r < 8; ++r) {
      int m = mr + r;
      if (m < M) {
        float v = acc[t][r] + bn + ex;
        if (EPI == EPI_RELU) v = fmaxf(v, 0.f);
        if (EPI == EPI_SOFTPLUS) v = (v > 20.f) ? v : log1pf(__expf(v));
        C[(size_t)m * ldc + n] = v;
      }
    }
  }
}

// ---------------------------------------------------------------- batchnorm

__global__ void k_bn_stats(const float* __restrict__ Z, float* __restrict__ sum,
                           float* __restrict__ sumsq) {
  int c = blockIdx.x * 64 + threadIdx.x;
  if (c >= EMB) return;
  float s = 0.f, s2 = 0.f;
  for (int r = blockIdx.y; r < N_NODES; r += gridDim.y) {
    float v = Z[(size_t)r * EMBP + c];
    s += v; s2 += v * v;
  }
  atomicAdd(&sum[c], s);
  atomicAdd(&sumsq[c], s2);
}

__global__ void k_bn_apply(const float* __restrict__ Z, const float* __restrict__ sum,
                           const float* __restrict__ sumsq, const float* __restrict__ g,
                           const float* __restrict__ b, float* __restrict__ H, int doRelu) {
  size_t idx = (size_t)blockIdx.x * blockDim.x + threadIdx.x;
  if (idx >= (size_t)N_NODES * EMBP) return;
  int c = (int)(idx % EMBP);
  float v = 0.f;
  if (c < EMB) {
    float mu = sum[c] * (1.f / N_NODES);
    float var = sumsq[c] * (1.f / N_NODES) - mu * mu;
    v = (Z[idx] - mu) * rsqrtf(var + EPSV) * g[c] + b[c];
    if (doRelu) v = fmaxf(v, 0.f);
  }
  H[idx] = v;
}

// ---------------------------------------------------------------- pooling / tail

__global__ void k_cnt(const int* __restrict__ batch, int* __restrict__ cnt) {
  int i = blockIdx.x * blockDim.x + threadIdx.x;
  if (i < N_NODES) atomicAdd(&cnt[batch[i]], 1);
}
__global__ void k_pool_sum(const int* __restrict__ batch, const float* __restrict__ H,
                           float* __restrict__ pool) {
  int i = blockIdx.x;
  int j = blockIdx.y * blockDim.x + threadIdx.x;
  if (j >= EMBP) return;
  atomicAdd(&pool[(size_t)batch[i] * EMBP + j], H[(size_t)i * EMBP + j]);
}
__global__ void k_pool_div(float* __restrict__ pool, const int* __restrict__ cnt) {
  size_t idx = (size_t)blockIdx.x * blockDim.x + threadIdx.x;
  if (idx >= (size_t)N_GRAPHS * EMBP) return;
  pool[idx] /= fmaxf((float)cnt[idx / EMBP], 1.f);
}

__global__ void k_qconst(const float* __restrict__ S, const float* __restrict__ qw,
                         const float* __restrict__ qb, float* __restrict__ qc) {
  int n = blockIdx.x * blockDim.x + threadIdx.x;
  if (n >= FEAT) return;
  float a = qb[n];
  for (int k = 0; k < FEAT; ++k) a += S[k] * qw[k * FEAT + n];
  qc[n] = a;
}

// dsum[g] = sum of dense rows (8 motif rows + hg row) via mol_idx scatter
__global__ void k_dsum_scatter(const float* __restrict__ motif, const float* __restrict__ hg,
                               const int* __restrict__ mol, float* __restrict__ dsum) {
  int r = blockIdx.x;  // 0 .. N_MOTIF + N_GRAPHS - 1
  int j = blockIdx.y * blockDim.x + threadIdx.x;
  if (j >= FEAT) return;
  int g = mol[r];
  float v = (r < N_MOTIF) ? motif[(size_t)r * FEAT + j]
                          : hg[(size_t)(r - N_MOTIF) * FEAT + j];
  atomicAdd(&dsum[(size_t)g * FEAT + j], v);
}

__global__ __launch_bounds__(256) void k_layernorm(float* __restrict__ X,
                                                   const float* __restrict__ g,
                                                   const float* __restrict__ b) {
  __shared__ float sm[256], sm2[256];
  int t = threadIdx.x;
  float* x = X + (size_t)blockIdx.x * FEAT;
  float v0 = x[t], v1 = x[t + 256];
  sm[t] = v0 + v1;
  sm2[t] = v0 * v0 + v1 * v1;
  __syncthreads();
  for (int s = 128; s > 0; s >>= 1) {
    if (t < s) { sm[t] += sm[t + s]; sm2[t] += sm2[t + s]; }
    __syncthreads();
  }
  float mu = sm[0] * (1.f / FEAT);
  float var = sm2[0] * (1.f / FEAT) - mu * mu;
  float rs = rsqrtf(var + EPSV);
  x[t]       = (v0 - mu) * rs * g[t] + b[t];
  x[t + 256] = (v1 - mu) * rs * g[t + 256] + b[t + 256];
}

__global__ void k_add(float* __restrict__ a, const float* __restrict__ b, size_t n) {
  size_t i = (size_t)blockIdx.x * blockDim.x + threadIdx.x;
  if (i < n) a[i] += b[i];
}

__global__ void k_hp2(const float* __restrict__ hg, const float* __restrict__ out,
                      bf16_t* __restrict__ dst) {
  size_t idx = (size_t)blockIdx.x * blockDim.x + threadIdx.x;
  if (idx >= (size_t)MPG * 1024) return;
  int r = (int)(idx / 1024), c = (int)(idx % 1024);
  float v = 0.f;
  if (r < N_GRAPHS)
    v = (c < FEAT) ? hg[(size_t)r * FEAT + c] : out[(size_t)r * FEAT + (c - FEAT)];
  dst[idx] = (bf16_t)v;
}

// ---------------------------------------------------------------- host side

static inline unsigned gblk(size_t n) { return (unsigned)((n + 255) / 256); }
static inline size_t shmB(int Kp) { return (size_t)64 * (Kp + LDSPAD) * sizeof(bf16_t); }

extern "C" void kernel_launch(void* const* d_in, const int* in_sizes, int n_in,
                              void* d_out, int out_size, void* d_ws, size_t ws_size,
                              hipStream_t stream) {
  (void)in_sizes; (void)n_in; (void)out_size; (void)ws_size;

  // ---- inputs (setup_inputs() dict insertion order) ----
  const int*   node_x  = (const int*)d_in[0];
  const int*   eidx    = (const int*)d_in[1];
  const int*   eattr   = (const int*)d_in[2];
  const int*   batch   = (const int*)d_in[3];
  const int*   mol_idx = (const int*)d_in[4];
  const float* motif   = (const float*)d_in[5];
  int pi = 6;
  const float* x_emb1 = (const float*)d_in[pi++];
  const float* x_emb2 = (const float*)d_in[pi++];
  struct Layer { const float *ee1, *ee2, *w1, *b1, *w2, *b2, *bng, *bnb; } L[NUM_LAYER];
  for (int l = 0; l < NUM_LAYER; ++l) {
    L[l].ee1 = (const float*)d_in[pi++]; L[l].ee2 = (const float*)d_in[pi++];
    L[l].w1  = (const float*)d_in[pi++]; L[l].b1  = (const float*)d_in[pi++];
    L[l].w2  = (const float*)d_in[pi++]; L[l].b2  = (const float*)d_in[pi++];
    L[l].bng = (const float*)d_in[pi++]; L[l].bnb = (const float*)d_in[pi++];
  }
  const float* feat_w = (const float*)d_in[pi++];
  const float* feat_b = (const float*)d_in[pi++];
  const float* Sv     = (const float*)d_in[pi++];
  const float* q_w = (const float*)d_in[pi++]; const float* q_b = (const float*)d_in[pi++];
  pi += 2;  // k_w, k_b: dead (softmax over singleton query axis -> A == 1)
  const float* v_w = (const float*)d_in[pi++]; const float* v_b = (const float*)d_in[pi++];
  const float* o_w = (const float*)d_in[pi++]; const float* o_b = (const float*)d_in[pi++];
  const float* ln0g = (const float*)d_in[pi++]; const float* ln0b = (const float*)d_in[pi++];
  const float* ln1g = (const float*)d_in[pi++]; const float* ln1b = (const float*)d_in[pi++];
  const float* p1_w = (const float*)d_in[pi++]; const float* p1_b = (const float*)d_in[pi++];
  const float* p2_w = (const float*)d_in[pi++]; const float* p2_b = (const float*)d_in[pi++];
  const float* p3_w = (const float*)d_in[pi++]; const float* p3_b = (const float*)d_in[pi++];

  // ---- workspace carve ----
  char* wp = (char*)d_ws;
  auto carve = [&](size_t bytes) -> void* {
    void* r = (void*)wp;
    wp += (bytes + 255) & ~(size_t)255;
    return r;
  };
  float*  H      = (float*)carve((size_t)MPNODES * EMBP * 4);
  float*  AGG    = (float*)carve((size_t)MPNODES * EMBP * 4);
  bf16_t* ABF    = (bf16_t*)carve((size_t)MPNODES * HIDP * 2);
  float*  Z      = (float*)carve((size_t)MPNODES * HIDP * 4);
  bf16_t* WT     = (bf16_t*)carve((size_t)262144 * 2);
  float*  STATS  = (float*)carve((size_t)2 * HIDP * 4);
  float*  POOL   = (float*)carve((size_t)MPG * EMBP * 4);
  int*    CNT    = (int*)carve((size_t)MPG * 4);
  bf16_t* POOLBF = (bf16_t*)carve((size_t)MPG * EMBP * 2);
  float*  QC     = (float*)carve((size_t)FEAT * 4);
  float*  DSUMF  = (float*)carve((size_t)N_GRAPHS * FEAT * 4);
  bf16_t* DSUMBF = (bf16_t*)carve((size_t)MPG * FEAT * 2);
  float*  OUTB   = (float*)carve((size_t)N_GRAPHS * FEAT * 4);
  bf16_t* OUTBF  = (bf16_t*)carve((size_t)MPG * FEAT * 2);
  float*  TMP    = (float*)carve((size_t)N_GRAPHS * FEAT * 4);
  bf16_t* HP2BF  = (bf16_t*)carve((size_t)MPG * 1024 * 2);
  float*  Z1     = (float*)carve((size_t)N_GRAPHS * 256 * 4);
  bf16_t* Z1BF   = (bf16_t*)carve((size_t)MPG * 256 * 2);
  float*  Z2     = (float*)carve((size_t)N_GRAPHS * 256 * 4);
  bf16_t* Z2BF   = (bf16_t*)carve((size_t)MPG * 256 * 2);

  float* hg   = (float*)d_out;                            // [5000 x 512]
  float* pred = (float*)d_out + (size_t)N_GRAPHS * FEAT;  // [5000 x 2]

  const dim3 B256(256);
  const size_t NE = (size_t)N_NODES * EMBP;

  // ---- node embedding ----
  k_embed<<<gblk(NE), B256, 0, stream>>>(node_x, x_emb1, x_emb2, H);

  // ---- GINE layers ----
  for (int l = 0; l < NUM_LAYER; ++l) {
    k_init_agg<<<gblk(NE), B256, 0, stream>>>(H, L[l].ee1, L[l].ee2, AGG);
    k_scatter<<<dim3(N_EDGES, 2), B256, 0, stream>>>(eidx, eattr, L[l].ee1, L[l].ee2, H, AGG);

    // z = relu(agg @ w1 + b1)
    k_to_bf16<<<gblk((size_t)MPNODES * EMBP), B256, 0, stream>>>(ABF, AGG, N_NODES, EMBP, EMBP, EMBP, MPNODES);
    k_wprep<<<gblk((size_t)HIDP * EMBP), B256, 0, stream>>>(WT, L[l].w1, EMB, HID, EMBP, HIDP);
    k_gemm_wmma<EPI_RELU><<<dim3(MPNODES / 128, HIDP / 64), B256, shmB(EMBP), stream>>>(
        ABF, EMBP, WT, EMBP, Z, HIDP, L[l].b1, 1.f, nullptr, N_NODES, HID);

    // z2 = z @ w2 + b2  (into AGG buffer, ld EMBP)
    k_to_bf16<<<gblk((size_t)MPNODES * HIDP), B256, 0, stream>>>(ABF, Z, N_NODES, HID, HIDP, HIDP, MPNODES);
    k_wprep<<<gblk((size_t)EMBP * HIDP), B256, 0, stream>>>(WT, L[l].w2, HID, EMB, HIDP, EMBP);
    k_gemm_wmma<EPI_NONE><<<dim3(MPNODES / 128, EMBP / 64), B256, shmB(HIDP), stream>>>(
        ABF, HIDP, WT, HIDP, AGG, EMBP, L[l].b2, 1.f, nullptr, N_NODES, EMB);

    // batchnorm (batch stats, biased var) + relu except last layer
    k_zero_f32<<<gblk(2 * HIDP), B256, 0, stream>>>(STATS, 2 * HIDP);
    k_bn_stats<<<dim3(5, 256), dim3(64), 0, stream>>>(AGG, STATS, STATS + HIDP);
    k_bn_apply<<<gblk(NE), B256, 0, stream>>>(AGG, STATS, STATS + HIDP, L[l].bng, L[l].bnb, H,
                                              l < NUM_LAYER - 1 ? 1 : 0);
  }

  // ---- mean pooling per graph ----
  k_zero_f32<<<gblk((size_t)MPG * EMBP), B256, 0, stream>>>(POOL, (size_t)MPG * EMBP);
  k_zero_i32<<<gblk(MPG), B256, 0, stream>>>(CNT, MPG);
  k_cnt<<<gblk(N_NODES), B256, 0, stream>>>(batch, CNT);
  k_pool_sum<<<dim3(N_NODES, 2), B256, 0, stream>>>(batch, H, POOL);
  k_pool_div<<<gblk((size_t)N_GRAPHS * EMBP), B256, 0, stream>>>(POOL, CNT);

  // hg = pool @ feat_w + feat_b  -> d_out
  k_to_bf16<<<gblk((size_t)MPG * EMBP), B256, 0, stream>>>(POOLBF, POOL, N_GRAPHS, EMBP, EMBP, EMBP, MPG);
  k_wprep<<<gblk((size_t)FEAT * EMBP), B256, 0, stream>>>(WT, feat_w, EMB, FEAT, EMBP, FEAT);
  k_gemm_wmma<EPI_NONE><<<dim3(MPG / 128, FEAT / 64), B256, shmB(EMBP), stream>>>(
      POOLBF, EMBP, WT, EMBP, hg, FEAT, feat_b, 1.f, nullptr, N_GRAPHS, FEAT);

  // ---- collapsed PMA: out = Qconst + (sum of dense rows) @ v_w + 9*v_b ----
  k_qconst<<<dim3(2), B256, 0, stream>>>(Sv, q_w, q_b, QC);
  k_zero_f32<<<gblk((size_t)N_GRAPHS * FEAT), B256, 0, stream>>>(DSUMF, (size_t)N_GRAPHS * FEAT);
  k_dsum_scatter<<<dim3(N_MOTIF + N_GRAPHS, 2), B256, 0, stream>>>(motif, hg, mol_idx, DSUMF);
  k_to_bf16<<<gblk((size_t)MPG * FEAT), B256, 0, stream>>>(DSUMBF, DSUMF, N_GRAPHS, FEAT, FEAT, FEAT, MPG);
  k_wprep<<<gblk((size_t)FEAT * FEAT), B256, 0, stream>>>(WT, v_w, FEAT, FEAT, FEAT, FEAT);
  k_gemm_wmma<EPI_NONE><<<dim3(MPG / 128, FEAT / 64), B256, shmB(FEAT), stream>>>(
      DSUMBF, FEAT, WT, FEAT, OUTB, FEAT, v_b, 9.f, QC, N_GRAPHS, FEAT);

  k_layernorm<<<dim3(N_GRAPHS), B256, 0, stream>>>(OUTB, ln0g, ln0b);

  // out = out + relu(out @ o_w + o_b)
  k_to_bf16<<<gblk((size_t)MPG * FEAT), B256, 0, stream>>>(OUTBF, OUTB, N_GRAPHS, FEAT, FEAT, FEAT, MPG);
  k_wprep<<<gblk((size_t)FEAT * FEAT), B256, 0, stream>>>(WT, o_w, FEAT, FEAT, FEAT, FEAT);
  k_gemm_wmma<EPI_RELU><<<dim3(MPG / 128, FEAT / 64), B256, shmB(FEAT), stream>>>(
      OUTBF, FEAT, WT, FEAT, TMP, FEAT, o_b, 1.f, nullptr, N_GRAPHS, FEAT);
  k_add<<<gblk((size_t)N_GRAPHS * FEAT), B256, 0, stream>>>(OUTB, TMP, (size_t)N_GRAPHS * FEAT);

  k_layernorm<<<dim3(N_GRAPHS), B256, 0, stream>>>(OUTB, ln1g, ln1b);

  // ---- prediction head ----
  k_hp2<<<gblk((size_t)MPG * 1024), B256, 0, stream>>>(hg, OUTB, HP2BF);
  k_wprep<<<gblk((size_t)256 * 1024), B256, 0, stream>>>(WT, p1_w, 1024, 256, 1024, 256);
  k_gemm_wmma<EPI_SOFTPLUS><<<dim3(MPG / 128, 256 / 64), B256, shmB(1024), stream>>>(
      HP2BF, 1024, WT, 1024, Z1, 256, p1_b, 1.f, nullptr, N_GRAPHS, 256);

  k_to_bf16<<<gblk((size_t)MPG * 256), B256, 0, stream>>>(Z1BF, Z1, N_GRAPHS, 256, 256, 256, MPG);
  k_wprep<<<gblk((size_t)256 * 256), B256, 0, stream>>>(WT, p2_w, 256, 256, 256, 256);
  k_gemm_wmma<EPI_SOFTPLUS><<<dim3(MPG / 128, 256 / 64), B256, shmB(256), stream>>>(
      Z1BF, 256, WT, 256, Z2, 256, p2_b, 1.f, nullptr, N_GRAPHS, 256);

  k_to_bf16<<<gblk((size_t)MPG * 256), B256, 0, stream>>>(Z2BF, Z2, N_GRAPHS, 256, 256, 256, MPG);
  k_wprep<<<gblk((size_t)64 * 256), B256, 0, stream>>>(WT, p3_w, 256, 2, 256, 64);
  k_gemm_wmma<EPI_NONE><<<dim3(MPG / 128, 1), B256, shmB(256), stream>>>(
      Z2BF, 256, WT, 256, pred, 2, p3_b, 1.f, nullptr, N_GRAPHS, 2);
}
